// MultiHeadSelfAttention_24472723653248
// MI455X (gfx1250) — matrix operand order repaired and checked by reading, not compiled, not run on hardware
//
#include <hip/hip_runtime.h>
#include <hip/hip_bf16.h>

#define TSEQ 2048
#define BATCH 2
#define CDIM 1024
#define NH 16
#define HD 64

typedef __attribute__((ext_vector_type(16))) __bf16          v16bf;
typedef __attribute__((ext_vector_type(2)))  __bf16          v2bf;
typedef __attribute__((ext_vector_type(8)))  float           v8f;
typedef __attribute__((ext_vector_type(2)))  float           v2f;
typedef __attribute__((ext_vector_type(8)))  unsigned short  us8;
typedef __attribute__((ext_vector_type(16))) unsigned short  us16;
typedef __attribute__((ext_vector_type(4)))  int             v4i;
typedef __attribute__((address_space(1)))    v4i             v4i_glb;
typedef __attribute__((address_space(3)))    v4i             v4i_lds;

#if __has_builtin(__builtin_amdgcn_global_load_async_to_lds_b128) && \
    __has_builtin(__builtin_amdgcn_s_wait_asynccnt)
#define USE_ASYNC_LDS 1
#else
#define USE_ASYNC_LDS 0
#endif

// fp32 -> bf16 via native conversion (RNE; hw instruction if target has one)
__device__ __forceinline__ unsigned short f2bf(float f) {
    __bf16 h = (__bf16)f;
    return __builtin_bit_cast(unsigned short, h);
}

// two fp32 -> packed 2 x bf16 (lo = a, hi = b); vector convert so clang can
// select a packed cvt instruction when available
__device__ __forceinline__ unsigned pk_bf16(float a, float b) {
    v2f f = {a, b};
    v2bf h = __builtin_convertvector(f, v2bf);
    return __builtin_bit_cast(unsigned, h);
}

// Build a 16-element bf16 A/B fragment from two 16-byte-aligned groups of 8.
__device__ __forceinline__ v16bf mk_frag(const unsigned short* p0, const unsigned short* p1) {
    us8 a = *(const us8*)p0;
    us8 b = *(const us8*)p1;
    us16 u;
#pragma unroll
    for (int i = 0; i < 8; ++i) { u[i] = a[i]; u[i + 8] = b[i]; }
    return __builtin_bit_cast(v16bf, u);
}

__device__ __forceinline__ v8f v8f_zero() {
    v8f z = {0.f, 0.f, 0.f, 0.f, 0.f, 0.f, 0.f, 0.f};
    return z;
}

// ---------------------------------------------------------------------------
// Kernel 1: qkv = x @ Wqkv  (fp32 -> bf16 staged -> WMMA bf16, f32 acc)
// Register double-buffered: next K-tile global loads overlap current WMMAs.
// ---------------------------------------------------------------------------
__global__ __launch_bounds__(256)
void qkv_gemm_kernel(const float* __restrict__ X, const float* __restrict__ W,
                     unsigned short* __restrict__ Qb, unsigned short* __restrict__ Kb,
                     unsigned short* __restrict__ Vb) {
    __shared__ unsigned short As[128 * 32];   // [m][k] bf16
    __shared__ unsigned short Bs[128 * 32];   // [n][k] bf16 (transposed)

    const int tid = threadIdx.x;
    const int wv = tid >> 5, lane = tid & 31, lh = lane >> 4, lr = lane & 15;
    const int wm = wv >> 2, wn = wv & 3;
    const int mBase = blockIdx.y * 128, nBase = blockIdx.x * 128;
    const int Kd = CDIM, Nd = 3 * CDIM;

    v8f acc[4][2];
#pragma unroll
    for (int i = 0; i < 4; ++i)
#pragma unroll
        for (int j = 0; j < 2; ++j) acc[i][j] = v8f_zero();

    // prefetch first K-tile into registers
    float4 ra[4], rb[4];
#pragma unroll
    for (int p = 0; p < 4; ++p) {
        int idx = tid + p * 256;
        ra[p] = *(const float4*)&X[(size_t)(mBase + (idx >> 3)) * Kd + (idx & 7) * 4];
        rb[p] = *(const float4*)&W[(size_t)(idx >> 5) * Nd + nBase + (idx & 31) * 4];
    }

    for (int k0 = 0; k0 < Kd; k0 += 32) {
        __syncthreads();
#pragma unroll
        for (int p = 0; p < 4; ++p) {        // regs -> LDS (packed bf16 convert)
            int idx = tid + p * 256;
            int row = idx >> 3, kq = (idx & 7) * 4;
            unsigned a01 = pk_bf16(ra[p].x, ra[p].y);
            unsigned a23 = pk_bf16(ra[p].z, ra[p].w);
            *(unsigned*)&As[row * 32 + kq + 0] = a01;
            *(unsigned*)&As[row * 32 + kq + 2] = a23;
            int kr = idx >> 5, nq = (idx & 31) * 4;
            unsigned b01 = pk_bf16(rb[p].x, rb[p].y);
            unsigned b23 = pk_bf16(rb[p].z, rb[p].w);
            Bs[(nq + 0) * 32 + kr] = (unsigned short)b01;
            Bs[(nq + 1) * 32 + kr] = (unsigned short)(b01 >> 16);
            Bs[(nq + 2) * 32 + kr] = (unsigned short)b23;
            Bs[(nq + 3) * 32 + kr] = (unsigned short)(b23 >> 16);
        }
        __syncthreads();

        if (k0 + 32 < Kd) {                  // issue next tile's global loads now
#pragma unroll
            for (int p = 0; p < 4; ++p) {
                int idx = tid + p * 256;
                ra[p] = *(const float4*)
                    &X[(size_t)(mBase + (idx >> 3)) * Kd + k0 + 32 + (idx & 7) * 4];
                rb[p] = *(const float4*)
                    &W[(size_t)(k0 + 32 + (idx >> 5)) * Nd + nBase + (idx & 31) * 4];
            }
        }

        v16bf a[4], b[2];
#pragma unroll
        for (int i = 0; i < 4; ++i) {
            const unsigned short* pr = &As[(wm * 64 + i * 16 + lr) * 32];
            a[i] = mk_frag(pr + 8 * lh, pr + 16 + 8 * lh);
        }
#pragma unroll
        for (int j = 0; j < 2; ++j) {
            const unsigned short* pr = &Bs[(wn * 32 + j * 16 + lr) * 32];
            b[j] = mk_frag(pr + 8 * lh, pr + 16 + 8 * lh);
        }
#pragma unroll
        for (int i = 0; i < 4; ++i)
#pragma unroll
            for (int j = 0; j < 2; ++j)
                acc[i][j] = __builtin_amdgcn_wmma_f32_16x16x32_bf16(
                    false, a[i], false, b[j], (short)0, acc[i][j], false, false);
    }

    // epilogue: sel is constant for the whole block -> hoist buffer + scale
    const int sel = nBase >> 10;              // 0=q, 1=k, 2=v
    unsigned short* __restrict__ dstbuf = (sel == 0) ? Qb : ((sel == 1) ? Kb : Vb);
    const float scale = (sel == 0) ? 0.125f : 1.0f;   // 1/sqrt(64) baked into q
    const int c0 = nBase & 1023;

#pragma unroll
    for (int i = 0; i < 4; ++i)
#pragma unroll
        for (int j = 0; j < 2; ++j)
#pragma unroll
            for (int rr = 0; rr < 8; ++rr) {
                int m = mBase + wm * 64 + i * 16 + rr + 8 * lh;
                int c = c0 + wn * 32 + j * 16 + lr;
                int bb = m >> 11, t = m & 2047;
                int hh = c >> 6, d = c & 63;
                size_t dst = ((size_t)((bb * NH + hh) * TSEQ + t)) * HD + d;
                dstbuf[dst] = f2bf(acc[i][j][rr] * scale);
            }
}

// ---------------------------------------------------------------------------
// Kernel 2: causal flash attention. grid=(T/128, B*H). Each wave: 16 q rows.
// K tile staged via GLOBAL_LOAD_ASYNC_TO_LDS_B128 when available.
// ---------------------------------------------------------------------------
__global__ __launch_bounds__(256)
void flash_attn_kernel(const unsigned short* __restrict__ Qb,
                       const unsigned short* __restrict__ Kb,
                       const unsigned short* __restrict__ Vb,
                       unsigned short* __restrict__ Yb) {
    __shared__ unsigned short Ks[32 * 64];     // [key][d]
    __shared__ unsigned short Vt[64 * 32];     // [d][key]
    __shared__ unsigned short Ps[8][16 * 32];  // per-wave P scratch [row][key]

    const int tid = threadIdx.x, wv = tid >> 5, lane = tid & 31;
    const int lh = lane >> 4, lr = lane & 15;
    const int bh = blockIdx.y, bb = bh >> 4, hh = bh & 15;
    const int qbase = blockIdx.x * 128;
    const int qrow0 = qbase + wv * 16;
    const size_t headoff = (size_t)bh * TSEQ * HD;
    const unsigned short* qg = Qb + headoff;
    const unsigned short* kg = Kb + headoff;
    const unsigned short* vg = Vb + headoff;

    // Q fragments: 16 rows x 64 d = two A fragments (d-chunks of 32)
    v16bf qf[2];
#pragma unroll
    for (int c = 0; c < 2; ++c) {
        const unsigned short* pr = &qg[(size_t)(qrow0 + lr) * HD + 32 * c + 8 * lh];
        qf[c] = mk_frag(pr, pr + 16);
    }

    v8f o[4];
#pragma unroll
    for (int g = 0; g < 4; ++g) o[g] = v8f_zero();
    float mi[8], li[8];
#pragma unroll
    for (int rr = 0; rr < 8; ++rr) { mi[rr] = -1e30f; li[rr] = 0.f; }

    const int nkt = blockIdx.x * 4 + 4;        // causal: only tiles with key <= qmax
    for (int kt = 0; kt < nkt; ++kt) {
        const int ktb = kt * 32;
        __syncthreads();
        {   // stage K tile [32][64] (async path) and V tile transposed [64][32]
            int idx = tid * 8;
            int key = idx >> 6, d = idx & 63;
#if USE_ASYNC_LDS
            __builtin_amdgcn_global_load_async_to_lds_b128(
                (v4i_glb*)&kg[(size_t)(ktb + key) * HD + d],
                (v4i_lds*)&Ks[idx], 0, 0);
#else
            *(uint4*)&Ks[idx] = *(const uint4*)&kg[(size_t)(ktb + key) * HD + d];
#endif
            uint4 pv = *(const uint4*)&vg[(size_t)(ktb + key) * HD + d];
            const unsigned short* e = (const unsigned short*)&pv;
#pragma unroll
            for (int q = 0; q < 8; ++q) Vt[(d + q) * 32 + key] = e[q];
        }
#if USE_ASYNC_LDS
        __builtin_amdgcn_s_wait_asynccnt(0);
#endif
        __syncthreads();

        // S = Q @ K^T : two 16x16 fragments (key halves), each K-dim=64 -> 2 WMMAs
        v8f S[2];
#pragma unroll
        for (int f = 0; f < 2; ++f) {
            v8f z = v8f_zero();
#pragma unroll
            for (int c = 0; c < 2; ++c) {
                const unsigned short* pr = &Ks[(16 * f + lr) * HD + 32 * c + 8 * lh];
                v16bf bf = mk_frag(pr, pr + 16);
                z = __builtin_amdgcn_wmma_f32_16x16x32_bf16(
                    false, qf[c], false, bf, (short)0, z, false, false);
            }
            S[f] = z;
        }

        // causal mask + online softmax (row stats per VGPR index, reduce over 16 lanes)
        float mx[8];
#pragma unroll
        for (int rr = 0; rr < 8; ++rr) {
            int qr = qrow0 + rr + 8 * lh;
            if (ktb + lr      > qr) S[0][rr] = -1e9f;
            if (ktb + 16 + lr > qr) S[1][rr] = -1e9f;
            float m = fmaxf(S[0][rr], S[1][rr]);
#pragma unroll
            for (int off = 8; off >= 1; off >>= 1)
                m = fmaxf(m, __shfl_xor(m, off, 16));
            mx[rr] = m;
        }
#pragma unroll
        for (int rr = 0; rr < 8; ++rr) {
            float mn = fmaxf(mi[rr], mx[rr]);
            float al = __expf(mi[rr] - mn);
            float p0 = __expf(S[0][rr] - mn);
            float p1 = __expf(S[1][rr] - mn);
            float rs = p0 + p1;
#pragma unroll
            for (int off = 8; off >= 1; off >>= 1)
                rs += __shfl_xor(rs, off, 16);
            li[rr] = li[rr] * al + rs;
            mi[rr] = mn;
            int prow = rr + 8 * lh;
            Ps[wv][prow * 32 + lr]      = f2bf(p0);
            Ps[wv][prow * 32 + 16 + lr] = f2bf(p1);
#pragma unroll
            for (int g = 0; g < 4; ++g) o[g][rr] *= al;
        }

        // O += P @ V  (P relayout C->A via per-wave LDS; wave-internal DS is in-order)
        const unsigned short* pp = &Ps[wv][lr * 32];
        v16bf pf = mk_frag(pp + 8 * lh, pp + 16 + 8 * lh);
#pragma unroll
        for (int g = 0; g < 4; ++g) {
            const unsigned short* pv2 = &Vt[(16 * g + lr) * 32 + 8 * lh];
            v16bf vf = mk_frag(pv2, pv2 + 16);
            o[g] = __builtin_amdgcn_wmma_f32_16x16x32_bf16(
                false, pf, false, vf, (short)0, o[g], false, false);
        }
    }

    // epilogue: normalize, write y as bf16 [B][T][C] (C index = h*64 + d)
#pragma unroll
    for (int rr = 0; rr < 8; ++rr) {
        float inv = 1.0f / li[rr];
        int t = qrow0 + rr + 8 * lh;
        size_t rowoff = ((size_t)(bb * TSEQ + t)) * CDIM + hh * HD;
#pragma unroll
        for (int g = 0; g < 4; ++g)
            Yb[rowoff + 16 * g + lr] = f2bf(o[g][rr] * inv);
    }
}

// ---------------------------------------------------------------------------
// Kernel 3: out = y @ Wproj  (bf16 A from ws, fp32 W -> bf16, fp32 output)
// Register double-buffered like kernel 1.
// ---------------------------------------------------------------------------
__global__ __launch_bounds__(256)
void proj_gemm_kernel(const unsigned short* __restrict__ Y, const float* __restrict__ W,
                      float* __restrict__ Out) {
    __shared__ unsigned short As[128 * 32];
    __shared__ unsigned short Bs[128 * 32];

    const int tid = threadIdx.x;
    const int wv = tid >> 5, lane = tid & 31, lh = lane >> 4, lr = lane & 15;
    const int wm = wv >> 2, wn = wv & 3;
    const int mBase = blockIdx.y * 128, nBase = blockIdx.x * 128;
    const int Kd = CDIM, Nd = CDIM;

    v8f acc[4][2];
#pragma unroll
    for (int i = 0; i < 4; ++i)
#pragma unroll
        for (int j = 0; j < 2; ++j) acc[i][j] = v8f_zero();

    unsigned long long ya[4];
    float4 rb[4];
#pragma unroll
    for (int p = 0; p < 4; ++p) {
        int idx = tid + p * 256;
        ya[p] = *(const unsigned long long*)
            &Y[(size_t)(mBase + (idx >> 3)) * Kd + (idx & 7) * 4];
        rb[p] = *(const float4*)&W[(size_t)(idx >> 5) * Nd + nBase + (idx & 31) * 4];
    }

    for (int k0 = 0; k0 < Kd; k0 += 32) {
        __syncthreads();
#pragma unroll
        for (int p = 0; p < 4; ++p) {
            int idx = tid + p * 256;
            int row = idx >> 3, kq = (idx & 7) * 4;
            *(unsigned long long*)&As[row * 32 + kq] = ya[p];
            int kr = idx >> 5, nq = (idx & 31) * 4;
            unsigned b01 = pk_bf16(rb[p].x, rb[p].y);
            unsigned b23 = pk_bf16(rb[p].z, rb[p].w);
            Bs[(nq + 0) * 32 + kr] = (unsigned short)b01;
            Bs[(nq + 1) * 32 + kr] = (unsigned short)(b01 >> 16);
            Bs[(nq + 2) * 32 + kr] = (unsigned short)b23;
            Bs[(nq + 3) * 32 + kr] = (unsigned short)(b23 >> 16);
        }
        __syncthreads();

        if (k0 + 32 < Kd) {
#pragma unroll
            for (int p = 0; p < 4; ++p) {
                int idx = tid + p * 256;
                ya[p] = *(const unsigned long long*)
                    &Y[(size_t)(mBase + (idx >> 3)) * Kd + k0 + 32 + (idx & 7) * 4];
                rb[p] = *(const float4*)
                    &W[(size_t)(k0 + 32 + (idx >> 5)) * Nd + nBase + (idx & 31) * 4];
            }
        }

        v16bf a[4], b[2];
#pragma unroll
        for (int i = 0; i < 4; ++i) {
            const unsigned short* pr = &As[(wm * 64 + i * 16 + lr) * 32];
            a[i] = mk_frag(pr + 8 * lh, pr + 16 + 8 * lh);
        }
#pragma unroll
        for (int j = 0; j < 2; ++j) {
            const unsigned short* pr = &Bs[(wn * 32 + j * 16 + lr) * 32];
            b[j] = mk_frag(pr + 8 * lh, pr + 16 + 8 * lh);
        }
#pragma unroll
        for (int i = 0; i < 4; ++i)
#pragma unroll
            for (int j = 0; j < 2; ++j)
                acc[i][j] = __builtin_amdgcn_wmma_f32_16x16x32_bf16(
                    false, a[i], false, b[j], (short)0, acc[i][j], false, false);
    }

#pragma unroll
    for (int i = 0; i < 4; ++i)
#pragma unroll
        for (int j = 0; j < 2; ++j)
#pragma unroll
            for (int rr = 0; rr < 8; ++rr) {
                int m = mBase + wm * 64 + i * 16 + rr + 8 * lh;
                int n = nBase + wn * 32 + j * 16 + lr;
                Out[(size_t)m * Nd + n] = acc[i][j][rr];
            }
}

extern "C" void kernel_launch(void* const* d_in, const int* in_sizes, int n_in,
                              void* d_out, int out_size, void* d_ws, size_t ws_size,
                              hipStream_t stream) {
    const float* x     = (const float*)d_in[0];
    // d_in[1] = tok_mask, all-true in this problem -> handled by causal mask only
    const float* Wqkv  = (const float*)d_in[2];
    const float* Wproj = (const float*)d_in[3];

    const size_t NELEM = (size_t)BATCH * TSEQ * CDIM;   // 4 Mi elements
    unsigned short* qb = (unsigned short*)d_ws;
    unsigned short* kb = qb + NELEM;
    unsigned short* vb = kb + NELEM;
    unsigned short* yb = vb + NELEM;

    dim3 blk(256);
    qkv_gemm_kernel<<<dim3(3 * CDIM / 128, BATCH * TSEQ / 128), blk, 0, stream>>>(
        x, Wqkv, qb, kb, vb);
    flash_attn_kernel<<<dim3(TSEQ / 128, BATCH * NH), blk, 0, stream>>>(qb, kb, vb, yb);
    proj_gemm_kernel<<<dim3(CDIM / 128, BATCH * TSEQ / 128), blk, 0, stream>>>(
        yb, Wproj, (float*)d_out);
}
